// DirectedDAGNN_16947940950525
// MI455X (gfx1250) — matrix operand order repaired and checked by reading, not compiled, not run on hardware
//
#include <hip/hip_runtime.h>
#include <math.h>

// ---------------------------------------------------------------------------
// CDNA5 (gfx1250) WMMA helpers: D = A(16x32 f16) * B(32x16 f16) + C(f32)
// ---------------------------------------------------------------------------
typedef __attribute__((ext_vector_type(16))) _Float16 v16h;
typedef __attribute__((ext_vector_type(8)))  float    v8f;
typedef __attribute__((ext_vector_type(4)))  _Float16 h4;

__device__ __forceinline__ v8f wmma_f16(v16h a, v16h b, v8f c) {
  // 8 args: (neg_a, A, neg_b, B, c_mod, C, reuse_a, reuse_b)
  return __builtin_amdgcn_wmma_f32_16x16x32_f16(false, a, false, b, (short)0, c,
                                                false, false);
}

// A fragment from an f16 row-major tile (LDS or global), top-left at (0, k0).
// Layout (wave32): lane l, m=l&15, hi=l>>4; elems 0..7 -> K=k0+hi*8+i,
// elems 8..15 -> K=k0+16+hi*8+i.
__device__ __forceinline__ v16h load_a_f16(const _Float16* A, int lda, int k0,
                                           int lane) {
  const int m = lane & 15, hi = lane >> 4;
  const _Float16* p = A + m * lda + k0 + hi * 8;
  v16h a;
#pragma unroll
  for (int i = 0; i < 8; ++i) {
    a[i]     = p[i];
    a[8 + i] = p[16 + i];
  }
  return a;
}

// B fragment for computing X @ W.T where W is [O x Kdim] f16 row-major.
// B[k][o] = W[o0+n][k]; lane l: n=l&15, hi=l>>4; elems 0..7 -> K=k0+hi*8+i,
// elems 8..15 -> K=k0+16+hi*8+i.
__device__ __forceinline__ v16h load_b_wt(const _Float16* W, int ldw, int o0,
                                          int k0, int lane) {
  const int n = lane & 15, hi = lane >> 4;
  const _Float16* p = W + (size_t)(o0 + n) * ldw + k0 + hi * 8;
  v16h b;
#pragma unroll
  for (int i = 0; i < 8; ++i) {
    b[i]     = p[i];
    b[8 + i] = p[16 + i];
  }
  return b;
}

template <int KD>
__device__ __forceinline__ v8f gemm_tile(const _Float16* A, int lda,
                                         const _Float16* W, int ldw, int o0,
                                         int lane) {
  v8f acc = {};
#pragma unroll
  for (int k0 = 0; k0 < KD; k0 += 32) {
    v16h a = load_a_f16(A, lda, k0, lane);
    v16h b = load_b_wt(W, ldw, o0, k0, lane);
    acc    = wmma_f16(a, b, acc);
  }
  return acc;
}

// ---------------------------------------------------------------------------
// Problem constants
// ---------------------------------------------------------------------------
#define NN 50000
#define EE 800000
#define HH 128
#define AHD 64
#define KK 10
#define OUTD 40
#define EPSV 1e-5f
#define ALPHA 0.1f

__device__ __forceinline__ void atomic_fadd(float* p, float v) {
  __hip_atomic_fetch_add(p, v, __ATOMIC_RELAXED, __HIP_MEMORY_SCOPE_AGENT);
}

// ---------------------------------------------------------------------------
// Weight conversion f32 -> f16 with zero row padding (flat indices)
// ---------------------------------------------------------------------------
__global__ void cvt_pad_kernel(_Float16* dst, const float* src, int n_src,
                               int n_dst) {
  int i = blockIdx.x * blockDim.x + threadIdx.x;
  if (i < n_dst) dst[i] = (i < n_src) ? (_Float16)src[i] : (_Float16)0.0f;
}

// ---------------------------------------------------------------------------
// Stem: base = relu(bn2(relu(bn1(x@W1.T+b1))@W2.T+b2)) + relu(bn1(...))
// One block = 16 rows, 8 waves, wave w owns output cols [16w,16w+16).
// ---------------------------------------------------------------------------
__global__ __launch_bounds__(256) void stem_kernel(
    const float* __restrict__ x, const _Float16* __restrict__ w1h,
    const _Float16* __restrict__ w2h, const float* __restrict__ lb1,
    const float* __restrict__ g1, const float* __restrict__ bb1,
    const float* __restrict__ m1, const float* __restrict__ v1,
    const float* __restrict__ lb2, const float* __restrict__ g2,
    const float* __restrict__ bb2, const float* __restrict__ m2,
    const float* __restrict__ v2, float* __restrict__ base) {
  __shared__ _Float16 sx[16 * HH];
  __shared__ _Float16 sh16[16 * HH];
  __shared__ float    sh32[16 * HH];

  const int row0 = blockIdx.x * 16;
  for (int i = threadIdx.x; i < 16 * HH; i += 256) {
    int r = i >> 7, c = i & 127;
    sx[i] = (_Float16)x[(size_t)(row0 + r) * HH + c];
  }
  __syncthreads();

  const int wave = threadIdx.x >> 5, lane = threadIdx.x & 31;
  const int o0 = wave * 16;
  const int n = lane & 15, hi = lane >> 4;
  const int ch = o0 + n;

  // GEMM1 + BN1 + ReLU
  v8f acc = gemm_tile<HH>(sx, HH, w1h, HH, o0, lane);
  {
    float s = rsqrtf(v1[ch] + EPSV) * g1[ch];
    float pre = lb1[ch] - m1[ch];
    float add = bb1[ch];
#pragma unroll
    for (int r = 0; r < 8; ++r) {
      int m = r + 8 * hi;
      float val = fmaxf((acc[r] + pre) * s + add, 0.0f);
      sh32[m * HH + ch] = val;
      sh16[m * HH + ch] = (_Float16)val;
    }
  }
  __syncthreads();

  // GEMM2 + BN2 + ReLU + residual
  v8f acc2 = gemm_tile<HH>(sh16, HH, w2h, HH, o0, lane);
  {
    float s = rsqrtf(v2[ch] + EPSV) * g2[ch];
    float pre = lb2[ch] - m2[ch];
    float add = bb2[ch];
#pragma unroll
    for (int r = 0; r < 8; ++r) {
      int m = r + 8 * hi;
      float val = fmaxf((acc2[r] + pre) * s + add, 0.0f) + sh32[m * HH + ch];
      base[(size_t)(row0 + m) * HH + ch] = val;
    }
  }
}

// ---------------------------------------------------------------------------
// Edge normalization
// ---------------------------------------------------------------------------
__global__ void zero_f32_kernel(float* p, int n) {
  int i = blockIdx.x * blockDim.x + threadIdx.x;
  if (i < n) p[i] = 0.0f;
}

__global__ void wsum_kernel(float* __restrict__ wsum,
                            const long long* __restrict__ row,
                            const float* __restrict__ ew, int e) {
  int i = blockIdx.x * blockDim.x + threadIdx.x;
  if (i < e) atomic_fadd(&wsum[(int)row[i]], ew[i]);
}

__global__ void enorm_kernel(float* __restrict__ en,
                             const long long* __restrict__ row,
                             const float* __restrict__ ew,
                             const float* __restrict__ wsum, int e) {
  int i = blockIdx.x * blockDim.x + threadIdx.x;
  if (i < e) en[i] = ew[i] / fmaxf(wsum[(int)row[i]], 1.0f);
}

// ---------------------------------------------------------------------------
// Propagation: nxt = ALPHA*base; nxt[col] += (1-ALPHA)*enorm*h[row]
// ---------------------------------------------------------------------------
__global__ void init_iter_kernel(float4* __restrict__ nxt,
                                 const float4* __restrict__ base, int n4) {
  int i = blockIdx.x * blockDim.x + threadIdx.x;
  if (i < n4) {
    float4 b = base[i];
    nxt[i] = make_float4(ALPHA * b.x, ALPHA * b.y, ALPHA * b.z, ALPHA * b.w);
  }
}

__global__ __launch_bounds__(256) void scatter_kernel(
    float* __restrict__ nxt, const float* __restrict__ cur,
    const long long* __restrict__ row, const long long* __restrict__ col,
    const float* __restrict__ en, int e) {
  int g = (blockIdx.x * blockDim.x + threadIdx.x) >> 5;  // wave per edge
  int lane = threadIdx.x & 31;
  if (g >= e) return;
  int r = (int)row[g];
  int c = (int)col[g];
  float w = (1.0f - ALPHA) * en[g];
  float4 hv = *(const float4*)(cur + (size_t)r * HH + lane * 4);
  float* dst = nxt + (size_t)c * HH + lane * 4;
  atomic_fadd(dst + 0, w * hv.x);
  atomic_fadd(dst + 1, w * hv.y);
  atomic_fadd(dst + 2, w * hv.z);
  atomic_fadd(dst + 3, w * hv.w);
}

// f32 state -> f16 snapshot (fast path)
__global__ void cvt_state_kernel(_Float16* __restrict__ dst,
                                 const float4* __restrict__ src, int n4) {
  int i = blockIdx.x * blockDim.x + threadIdx.x;
  if (i < n4) {
    float4 v = src[i];
    h4 o;
    o[0] = (_Float16)v.x;
    o[1] = (_Float16)v.y;
    o[2] = (_Float16)v.z;
    o[3] = (_Float16)v.w;
    *(h4*)(dst + (size_t)i * 4) = o;
  }
}

// ---------------------------------------------------------------------------
// Attention: w = softmax(gelu([base|hK]@A1.T+b1)@A2.T+b2)  -> attw[N][16]
// One block = 16 rows, 4 waves.
// ---------------------------------------------------------------------------
__global__ __launch_bounds__(128) void attention_kernel(
    const float* __restrict__ base, const float* __restrict__ hfin,
    const _Float16* __restrict__ a1h, const float* __restrict__ a1b,
    const _Float16* __restrict__ a2h, const float* __restrict__ a2b,
    float* __restrict__ attw) {
  __shared__ _Float16 sctx[16 * 256];
  __shared__ _Float16 sa1[16 * AHD];
  __shared__ float    ss[16 * 16];

  const int row0 = blockIdx.x * 16;
  for (int i = threadIdx.x; i < 16 * 256; i += 128) {
    int r = i >> 8, c = i & 255;
    float v = (c < HH) ? base[(size_t)(row0 + r) * HH + c]
                       : hfin[(size_t)(row0 + r) * HH + (c - HH)];
    sctx[i] = (_Float16)v;
  }
  __syncthreads();

  const int wave = threadIdx.x >> 5, lane = threadIdx.x & 31;
  const int n = lane & 15, hi = lane >> 4;

  // layer 1: [16x256] @ [256x64] -> gelu -> sa1
  {
    const int o0 = wave * 16;
    const int ch = o0 + n;
    v8f acc = gemm_tile<256>(sctx, 256, a1h, 256, o0, lane);
    float bias = a1b[ch];
#pragma unroll
    for (int r = 0; r < 8; ++r) {
      int m = r + 8 * hi;
      float y = acc[r] + bias;
      float g = 0.5f * y * (1.0f + erff(y * 0.70710678118654752f));
      sa1[m * AHD + ch] = (_Float16)g;
    }
  }
  __syncthreads();

  // layer 2: [16x64] @ [64x16(pad of 11)] -> ss
  if (wave == 0) {
    v8f acc = gemm_tile<AHD>(sa1, AHD, a2h, AHD, 0, lane);
    float bias = (n < (KK + 1)) ? a2b[n] : 0.0f;
#pragma unroll
    for (int r = 0; r < 8; ++r) {
      int m = r + 8 * hi;
      ss[m * 16 + n] = acc[r] + bias;
    }
  }
  __syncthreads();

  if (threadIdx.x < 16) {
    int m = threadIdx.x;
    float mx = -3.0e38f;
#pragma unroll
    for (int j = 0; j <= KK; ++j) mx = fmaxf(mx, ss[m * 16 + j]);
    float e[KK + 1];
    float sum = 0.0f;
#pragma unroll
    for (int j = 0; j <= KK; ++j) {
      e[j] = __expf(ss[m * 16 + j] - mx);
      sum += e[j];
    }
    float inv = 1.0f / sum;
#pragma unroll
    for (int j = 0; j <= KK; ++j)
      attw[(size_t)(row0 + m) * 16 + j] = e[j] * inv;
  }
}

// ---------------------------------------------------------------------------
// Fast path: fused = sum_k attw[:,k] * x_k  (x0=base f32, x10=hfin f32,
// x1..x9 stored f16 at xs + (k-1)*N*H)
// ---------------------------------------------------------------------------
__global__ void fused_combine_kernel(float4* __restrict__ fused,
                                     const float* __restrict__ attw,
                                     const float4* __restrict__ base,
                                     const float4* __restrict__ hfin,
                                     const _Float16* __restrict__ xs, int n4) {
  int i = blockIdx.x * blockDim.x + threadIdx.x;
  if (i >= n4) return;
  const float* w = attw + (size_t)(i >> 5) * 16;
  float4 b = base[i];
  float4 hf = hfin[i];
  float w0 = w[0], wK = w[KK];
  float fx = w0 * b.x + wK * hf.x;
  float fy = w0 * b.y + wK * hf.y;
  float fz = w0 * b.z + wK * hf.z;
  float fw = w0 * b.w + wK * hf.w;
#pragma unroll
  for (int k = 1; k <= KK - 1; ++k) {
    h4 hv = *(const h4*)(xs + (size_t)(k - 1) * (size_t)NN * HH +
                         (size_t)i * 4);
    float wk = w[k];
    fx += wk * (float)hv[0];
    fy += wk * (float)hv[1];
    fz += wk * (float)hv[2];
    fw += wk * (float)hv[3];
  }
  fused[i] = make_float4(fx, fy, fz, fw);
}

// ---------------------------------------------------------------------------
// Slow path helpers: fused (+)= attw[node][k] * h
// ---------------------------------------------------------------------------
__global__ void fused_set_kernel(float4* __restrict__ fused,
                                 const float* __restrict__ attw,
                                 const float4* __restrict__ h, int n4) {
  int i = blockIdx.x * blockDim.x + threadIdx.x;
  if (i < n4) {
    float w = attw[(size_t)(i >> 5) * 16 + 0];
    float4 hv = h[i];
    fused[i] = make_float4(w * hv.x, w * hv.y, w * hv.z, w * hv.w);
  }
}

__global__ void fused_acc_kernel(float4* __restrict__ fused,
                                 const float* __restrict__ attw,
                                 const float4* __restrict__ h, int k, int n4) {
  int i = blockIdx.x * blockDim.x + threadIdx.x;
  if (i < n4) {
    float w = attw[(size_t)(i >> 5) * 16 + k];
    float4 hv = h[i];
    float4 f = fused[i];
    f.x += w * hv.x;
    f.y += w * hv.y;
    f.z += w * hv.z;
    f.w += w * hv.w;
    fused[i] = f;
  }
}

// ---------------------------------------------------------------------------
// Head: out = relu(bn3(fused@H1.T+b1)) @ H2.T + b2
// One block = 16 rows, 4 waves.
// ---------------------------------------------------------------------------
__global__ __launch_bounds__(128) void head_kernel(
    const float* __restrict__ fused, const _Float16* __restrict__ h1h,
    const float* __restrict__ h1b, const float* __restrict__ g3,
    const float* __restrict__ b3, const float* __restrict__ m3,
    const float* __restrict__ v3, const _Float16* __restrict__ h2h,
    const float* __restrict__ h2b, float* __restrict__ out) {
  __shared__ _Float16 sf[16 * HH];
  __shared__ _Float16 sz[16 * AHD];

  const int row0 = blockIdx.x * 16;
  for (int i = threadIdx.x; i < 16 * HH; i += 128) {
    int r = i >> 7, c = i & 127;
    sf[i] = (_Float16)fused[(size_t)(row0 + r) * HH + c];
  }
  __syncthreads();

  const int wave = threadIdx.x >> 5, lane = threadIdx.x & 31;
  const int n = lane & 15, hi = lane >> 4;

  // head1 + BN3 + ReLU
  {
    const int o0 = wave * 16;
    const int ch = o0 + n;
    v8f acc = gemm_tile<HH>(sf, HH, h1h, HH, o0, lane);
    float s = rsqrtf(v3[ch] + EPSV) * g3[ch];
    float pre = h1b[ch] - m3[ch];
    float add = b3[ch];
#pragma unroll
    for (int r = 0; r < 8; ++r) {
      int m = r + 8 * hi;
      float val = fmaxf((acc[r] + pre) * s + add, 0.0f);
      sz[m * AHD + ch] = (_Float16)val;
    }
  }
  __syncthreads();

  // head2: [16x64] @ [64x48(pad of 40)]
  if (wave < 3) {
    const int o0 = wave * 16;
    const int col = o0 + n;
    v8f acc = gemm_tile<AHD>(sz, AHD, h2h, AHD, o0, lane);
    if (col < OUTD) {
      float bias = h2b[col];
#pragma unroll
      for (int r = 0; r < 8; ++r) {
        int m = r + 8 * hi;
        out[(size_t)(row0 + m) * OUTD + col] = acc[r] + bias;
      }
    }
  }
}

// ---------------------------------------------------------------------------
// Host launcher
// ---------------------------------------------------------------------------
extern "C" void kernel_launch(void* const* d_in, const int* in_sizes, int n_in,
                              void* d_out, int out_size, void* d_ws,
                              size_t ws_size, hipStream_t stream) {
  const float*     x      = (const float*)d_in[0];
  const long long* ei     = (const long long*)d_in[1];  // [2, E] int64
  const float*     ew     = (const float*)d_in[2];
  const float*     lin1_w = (const float*)d_in[3];
  const float*     lin1_b = (const float*)d_in[4];
  const float*     bn1_g  = (const float*)d_in[5];
  const float*     bn1_b  = (const float*)d_in[6];
  const float*     bn1_m  = (const float*)d_in[7];
  const float*     bn1_v  = (const float*)d_in[8];
  const float*     lin2_w = (const float*)d_in[9];
  const float*     lin2_b = (const float*)d_in[10];
  const float*     bn2_g  = (const float*)d_in[11];
  const float*     bn2_b  = (const float*)d_in[12];
  const float*     bn2_m  = (const float*)d_in[13];
  const float*     bn2_v  = (const float*)d_in[14];
  const float*     att1_w = (const float*)d_in[15];
  const float*     att1_b = (const float*)d_in[16];
  const float*     att2_w = (const float*)d_in[17];
  const float*     att2_b = (const float*)d_in[18];
  const float*     head1_w = (const float*)d_in[19];
  const float*     head1_b = (const float*)d_in[20];
  const float*     bn3_g  = (const float*)d_in[21];
  const float*     bn3_b  = (const float*)d_in[22];
  const float*     bn3_m  = (const float*)d_in[23];
  const float*     bn3_v  = (const float*)d_in[24];
  const float*     head2_w = (const float*)d_in[25];
  const float*     head2_b = (const float*)d_in[26];
  float* out = (float*)d_out;

  const long long* row = ei;
  const long long* col = ei + EE;

  // workspace carve-out (256B aligned)
  size_t off = 0;
  auto alloc = [&](size_t bytes) -> void* {
    void* p = (char*)d_ws + off;
    off += (bytes + 255) & ~(size_t)255;
    return p;
  };
  float* base  = (float*)alloc((size_t)NN * HH * 4);
  float* bufA  = (float*)alloc((size_t)NN * HH * 4);
  float* bufB  = (float*)alloc((size_t)NN * HH * 4);
  float* fused = (float*)alloc((size_t)NN * HH * 4);
  float* attw  = (float*)alloc((size_t)NN * 16 * 4);
  float* wsum  = (float*)alloc((size_t)NN * 4);
  float* en    = (float*)alloc((size_t)EE * 4);
  _Float16* w1h = (_Float16*)alloc(HH * HH * 2);
  _Float16* w2h = (_Float16*)alloc(HH * HH * 2);
  _Float16* a1h = (_Float16*)alloc(AHD * 2 * HH * 2);
  _Float16* a2h = (_Float16*)alloc(16 * AHD * 2);   // 11 rows padded to 16
  _Float16* h1h = (_Float16*)alloc(AHD * HH * 2);
  _Float16* h2h = (_Float16*)alloc(48 * AHD * 2);   // 40 rows padded to 48

  // Fast path: snapshot x_1..x_{K-1} as f16 if workspace allows.
  const size_t xs_bytes = (size_t)(KK - 1) * (size_t)NN * HH * 2;
  _Float16* xs = nullptr;
  if (off + xs_bytes <= ws_size) xs = (_Float16*)alloc(xs_bytes);

  const int NH  = NN * HH;
  const int NH4 = NH / 4;

  // 1) convert weights to f16 (with zero-row padding where needed)
  auto cvt = [&](_Float16* dst, const float* src, int nsrc, int ndst) {
    cvt_pad_kernel<<<(ndst + 255) / 256, 256, 0, stream>>>(dst, src, nsrc, ndst);
  };
  cvt(w1h, lin1_w, HH * HH, HH * HH);
  cvt(w2h, lin2_w, HH * HH, HH * HH);
  cvt(a1h, att1_w, AHD * 2 * HH, AHD * 2 * HH);
  cvt(a2h, att2_w, (KK + 1) * AHD, 16 * AHD);
  cvt(h1h, head1_w, AHD * HH, AHD * HH);
  cvt(h2h, head2_w, OUTD * AHD, 48 * AHD);

  // 2) stem -> base (== x_0)
  stem_kernel<<<NN / 16, 256, 0, stream>>>(x, w1h, w2h, lin1_b, bn1_g, bn1_b,
                                           bn1_m, bn1_v, lin2_b, bn2_g, bn2_b,
                                           bn2_m, bn2_v, base);

  // 3) edge normalization
  zero_f32_kernel<<<(NN + 255) / 256, 256, 0, stream>>>(wsum, NN);
  wsum_kernel<<<(EE + 255) / 256, 256, 0, stream>>>(wsum, row, ew, EE);
  enorm_kernel<<<(EE + 255) / 256, 256, 0, stream>>>(en, row, ew, wsum, EE);

  const int gridScatter = (EE + 7) / 8;  // wave (32 lanes) per edge, 8/block
  const int gridNH4 = (NH4 + 255) / 256;

  if (xs) {
    // ---- single propagation pass, snapshotting intermediate states ----
    const float* cur = base;
    for (int k = 1; k <= KK; ++k) {
      float* nxt = (k & 1) ? bufA : bufB;
      init_iter_kernel<<<gridNH4, 256, 0, stream>>>((float4*)nxt,
                                                    (const float4*)base, NH4);
      scatter_kernel<<<gridScatter, 256, 0, stream>>>(nxt, cur, row, col, en,
                                                      EE);
      if (k < KK)
        cvt_state_kernel<<<gridNH4, 256, 0, stream>>>(
            xs + (size_t)(k - 1) * NH, (const float4*)nxt, NH4);
      cur = nxt;
    }
    const float* hfin = cur;  // x_K
    attention_kernel<<<NN / 16, 128, 0, stream>>>(base, hfin, a1h, att1_b, a2h,
                                                  att2_b, attw);
    fused_combine_kernel<<<gridNH4, 256, 0, stream>>>(
        (float4*)fused, attw, (const float4*)base, (const float4*)hfin, xs,
        NH4);
  } else {
    // ---- two-pass recompute (bounded workspace) ----
    const float* cur = base;
    for (int k = 1; k <= KK; ++k) {
      float* nxt = (k & 1) ? bufA : bufB;
      init_iter_kernel<<<gridNH4, 256, 0, stream>>>((float4*)nxt,
                                                    (const float4*)base, NH4);
      scatter_kernel<<<gridScatter, 256, 0, stream>>>(nxt, cur, row, col, en,
                                                      EE);
      cur = nxt;
    }
    const float* hfin = cur;  // x_K
    attention_kernel<<<NN / 16, 128, 0, stream>>>(base, hfin, a1h, att1_b, a2h,
                                                  att2_b, attw);
    fused_set_kernel<<<gridNH4, 256, 0, stream>>>((float4*)fused, attw,
                                                  (const float4*)base, NH4);
    cur = base;
    for (int k = 1; k <= KK; ++k) {
      float* nxt = (k & 1) ? bufA : bufB;
      init_iter_kernel<<<gridNH4, 256, 0, stream>>>((float4*)nxt,
                                                    (const float4*)base, NH4);
      scatter_kernel<<<gridScatter, 256, 0, stream>>>(nxt, cur, row, col, en,
                                                      EE);
      fused_acc_kernel<<<gridNH4, 256, 0, stream>>>((float4*)fused, attw,
                                                    (const float4*)nxt, k, NH4);
      cur = nxt;
    }
  }

  // 7) head -> out
  head_kernel<<<NN / 16, 128, 0, stream>>>(fused, h1h, head1_b, bn3_g, bn3_b,
                                           bn3_m, bn3_v, h2h, head2_b, out);
}
